// FeedforwardLanguageModel_26774826123582
// MI455X (gfx1250) — compile-verified, matrix operand order checked by reading, not run on hardware
//
#include <hip/hip_runtime.h>
#include <hip/hip_bf16.h>
#include <stdint.h>

// ---- problem dims ----
#define T_DIM 128
#define B_DIM 16
#define V_DIM 50000
#define D_DIM 300
#define H_DIM 256
#define WIN   5
#define M_ROWS (T_DIM * B_DIM)      // 2048
#define K1_REAL (2 * WIN * D_DIM)   // 3000
#define K1      3072                // padded to multiple of 32
#define KB1     (K1 / 32)           // 96 k-steps in GEMM1
#define KB2     (H_DIM / 32)        // 8 k-steps in GEMM2
#define NT1     (H_DIM / 16)        // 16 n-subtiles in GEMM1
#define NT2     (V_DIM / 16)        // 3125 n-subtiles in GEMM2

// LDS row pads (bf16 elems) so row stride mod 64 banks == 4 DWORDs -> conflict-free ds_load_b128
#define XS_STRIDE 1032              // kernel1: 1024 + 8
#define HS_STRIDE 264               // kernel2: 256 + 8

typedef __attribute__((ext_vector_type(16))) __bf16 v16bf;
typedef __attribute__((ext_vector_type(8)))  float  v8f;

// float -> bf16 (round to nearest even), stored as ushort
__device__ __forceinline__ unsigned short f2bf(float f) {
    union { float f; unsigned u; } c; c.f = f;
    unsigned u = c.u;
    u += 0x7FFFu + ((u >> 16) & 1u);
    return (unsigned short)(u >> 16);
}

// Fragment-contiguous layout: fragment f = 1KB; lane L holds its 16 bf16 at f*512 + L*16.
// Element order inside lane matches ISA 16-bit A/B VGPR layout:
//   lanes 0-15 : K {base+0..7, base+16..23};  lanes 16-31 : K {base+8..15, base+24..31}
__device__ __forceinline__ v16bf load_frag_f(const unsigned short* __restrict__ Wf,
                                             int fragIdx, int lane) {
    const unsigned short* p = Wf + ((size_t)fragIdx * 32 + lane) * 16;
    union { uint4 u4[2]; v16bf v; } r;
    r.u4[0] = *(const uint4*)(p);
    r.u4[1] = *(const uint4*)(p + 8);
    return r.v;
}

// A fragment from padded row-major LDS tile (stride in bf16 elems)
__device__ __forceinline__ v16bf load_frag_lds(const unsigned short* xs, int row0,
                                               int stride, int k0, int lane) {
    const unsigned short* p = xs + (row0 + (lane & 15)) * stride + k0 + ((lane & 16) ? 8 : 0);
    union { uint4 u4[2]; v16bf v; } r;
    r.u4[0] = *(const uint4*)(p);
    r.u4[1] = *(const uint4*)(p + 16);
    return r.v;
}

// A fragment from row-major global bf16 [rows, stride]
__device__ __forceinline__ v16bf load_frag_g(const unsigned short* __restrict__ base,
                                             int row, int stride, int k0, int lane) {
    const unsigned short* p = base + (size_t)row * stride + k0 + ((lane & 16) ? 8 : 0);
    union { uint4 u4[2]; v16bf v; } r;
    r.u4[0] = *(const uint4*)(p);
    r.u4[1] = *(const uint4*)(p + 16);
    return r.v;
}

// ---- W2 [256,50000] f32 -> fragment-layout bf16 W2f (3125 x 8 fragments) ----
// one thread per 8-elem half-chunk: 16B contiguous write; reads coalesce to 64B segments
__global__ __launch_bounds__(256) void k_conv_w2(const float* __restrict__ W2,
                                                 unsigned short* __restrict__ W2f) {
    int q = blockIdx.x * 256 + threadIdx.x;      // 0 .. NT2*KB2*64-1  (1,600,000)
    int c  = q & 1;
    int L  = (q >> 1) & 31;
    int f  = q >> 6;
    int kb = f & 7;
    int nt = f >> 3;
    int n     = nt * 16 + (L & 15);
    int kbase = kb * 32 + ((L & 16) ? 8 : 0) + c * 16;
    unsigned short* o = W2f + (size_t)q * 8;
#pragma unroll
    for (int e = 0; e < 8; ++e)
        o[e] = f2bf(W2[(size_t)(kbase + e) * V_DIM + n]);
}

// ---- W1 [3000,256] f32 -> fragment-layout bf16 W1f (16 x 96 fragments), K zero-padded ----
__global__ __launch_bounds__(256) void k_conv_w1(const float* __restrict__ W1,
                                                 unsigned short* __restrict__ W1f) {
    int q = blockIdx.x * 256 + threadIdx.x;      // 0 .. NT1*KB1*64-1  (98,304)
    int c  = q & 1;
    int L  = (q >> 1) & 31;
    int f  = q >> 6;
    int kb = f % KB1;
    int nt = f / KB1;
    int n     = nt * 16 + (L & 15);
    int kbase = kb * 32 + ((L & 16) ? 8 : 0) + c * 16;
    unsigned short* o = W1f + (size_t)q * 8;
#pragma unroll
    for (int e = 0; e < 8; ++e) {
        int k = kbase + e;
        o[e] = (k < K1_REAL) ? f2bf(W1[(size_t)k * H_DIM + n]) : (unsigned short)0;
    }
}

// ---- kernel 1: embedding gather + GEMM1 + tanh -> h_bf [2048,256] bf16 ----
// One block per t (the 16 M-rows are the 16 batch entries of step t).
__global__ __launch_bounds__(256) void k_embed_gemm1(const int* __restrict__ text,
                                                     const float* __restrict__ Ef,
                                                     const float* __restrict__ Eu,
                                                     const unsigned short* __restrict__ W1f,
                                                     const float* __restrict__ b1,
                                                     unsigned short* __restrict__ h_bf) {
    __shared__ unsigned short xs[16 * XS_STRIDE];  // ~33KB, padded rows
    const int tid  = threadIdx.x;
    const int lane = tid & 31;
    const int wave = tid >> 5;                     // 0..7
    const int t    = blockIdx.x;                   // 0..127
    const int nt0  = wave * 2;                     // wave owns n-subtiles nt0, nt0+1

    v8f acc0 = {};
    v8f acc1 = {};

    for (int s = 0; s < 3; ++s) {                  // 3 chunks of 1024 K
        __syncthreads();
        for (int i = tid; i < 16 * 1024; i += 256) {
            int b  = i >> 10;
            int kk = i & 1023;
            int kg = s * 1024 + kk;
            float v = 0.f;
            if (kg < K1_REAL) {
                int w  = kg / (2 * D_DIM);
                int j  = kg - w * (2 * D_DIM);
                int tt = t - WIN + w;
                int tok = (tt >= 0) ? text[tt * B_DIM + b] : 0;
                v = (j < D_DIM) ? Ef[(size_t)tok * D_DIM + j]
                                : Eu[(size_t)tok * D_DIM + (j - D_DIM)];
            }
            xs[b * XS_STRIDE + kk] = f2bf(v);
        }
        __syncthreads();

        for (int kb = 0; kb < 32; ++kb) {
            int kbg = s * 32 + kb;                 // global k-step
            v16bf a  = load_frag_lds(xs, 0, XS_STRIDE, kb * 32, lane);
            v16bf b0 = load_frag_f(W1f, nt0 * KB1 + kbg, lane);
            acc0 = __builtin_amdgcn_wmma_f32_16x16x32_bf16(false, a, false, b0,
                                                           (short)0, acc0, false, false);
            v16bf b1f = load_frag_f(W1f, (nt0 + 1) * KB1 + kbg, lane);
            acc1 = __builtin_amdgcn_wmma_f32_16x16x32_bf16(false, a, false, b1f,
                                                           (short)0, acc1, false, false);
        }
    }

    const float bias0 = b1[nt0 * 16 + (lane & 15)];
    const float bias1 = b1[nt0 * 16 + 16 + (lane & 15)];
    const int mrow = t * 16 + ((lane & 16) ? 8 : 0);
#pragma unroll
    for (int r = 0; r < 8; ++r) {
        int m = mrow + r;
        h_bf[m * H_DIM + nt0 * 16 + (lane & 15)]      = f2bf(tanhf(acc0[r] + bias0));
        h_bf[m * H_DIM + nt0 * 16 + 16 + (lane & 15)] = f2bf(tanhf(acc1[r] + bias1));
    }
}

// ---- kernel 2: logits = h @ W2 + b2 ----
// Block tile M=64 (h strip staged once in padded LDS), 8 waves x 4 N-subtiles each.
// B fragments: single 1KB-contiguous stream per (nt,kb) from L2-resident W2f.
// Output stores nontemporal: 410MB write-once stream must not evict W2f from L2.
__global__ __launch_bounds__(256) void k_gemm2(const unsigned short* __restrict__ h_bf,
                                               const unsigned short* __restrict__ W2f,
                                               const float* __restrict__ b2,
                                               float* __restrict__ out) {
    __shared__ unsigned short hs[64 * HS_STRIDE];  // ~33KB padded
    const int tid  = threadIdx.x;
    const int lane = tid & 31;
    const int wave = tid >> 5;
    const int m0   = blockIdx.y * 64;

    for (int i = tid; i < 64 * 256; i += 256) {
        int row = i >> 8, col = i & 255;
        hs[row * HS_STRIDE + col] = h_bf[(m0 + row) * H_DIM + col];
    }
    __syncthreads();

#pragma unroll
    for (int s = 0; s < 4; ++s) {
        int nt = blockIdx.x * 32 + wave * 4 + s;   // wave-uniform guard: EXEC all-ones at WMMA
        if (nt >= NT2) continue;
        const float bias = b2[nt * 16 + (lane & 15)];
        v8f acc[4];
#pragma unroll
        for (int ms = 0; ms < 4; ++ms)
#pragma unroll
            for (int r = 0; r < 8; ++r)
                acc[ms][r] = bias;

#pragma unroll
        for (int kb = 0; kb < KB2; ++kb) {
            v16bf b = load_frag_f(W2f, nt * KB2 + kb, lane);
#pragma unroll
            for (int ms = 0; ms < 4; ++ms) {
                v16bf a = load_frag_lds(hs, ms * 16, HS_STRIDE, kb * 32, lane);
                acc[ms] = __builtin_amdgcn_wmma_f32_16x16x32_bf16(false, a, false, b,
                                                                  (short)0, acc[ms], false, false);
            }
        }

        const int n = nt * 16 + (lane & 15);
        const int mb = m0 + ((lane & 16) ? 8 : 0);
#pragma unroll
        for (int ms = 0; ms < 4; ++ms)
#pragma unroll
            for (int r = 0; r < 8; ++r)
                __builtin_nontemporal_store(acc[ms][r],
                    &out[(size_t)(mb + ms * 16 + r) * V_DIM + n]);
    }
}

extern "C" void kernel_launch(void* const* d_in, const int* in_sizes, int n_in,
                              void* d_out, int out_size, void* d_ws, size_t ws_size,
                              hipStream_t stream) {
    (void)in_sizes; (void)n_in; (void)out_size; (void)ws_size;
    const int*   text = (const int*)d_in[0];
    const float* Ef   = (const float*)d_in[1];
    const float* Eu   = (const float*)d_in[2];
    const float* W1   = (const float*)d_in[3];
    const float* b1   = (const float*)d_in[4];
    const float* W2   = (const float*)d_in[5];
    const float* b2   = (const float*)d_in[6];
    float* out = (float*)d_out;

    char* ws = (char*)d_ws;
    unsigned short* W2f = (unsigned short*)(ws);                        // 3125*8*1KB = 25,600,000 B
    unsigned short* W1f = (unsigned short*)(ws + 25600000);             // 16*96*1KB  =  1,572,864 B
    unsigned short* hbf = (unsigned short*)(ws + 25600000 + 1572864);   // 2048*256*2 =  1,048,576 B

    k_conv_w2<<<(NT2 * KB2 * 64) / 256, 256, 0, stream>>>(W2, W2f);
    k_conv_w1<<<(NT1 * KB1 * 64) / 256, 256, 0, stream>>>(W1, W1f);
    k_embed_gemm1<<<T_DIM, 256, 0, stream>>>(text, Ef, Eu, W1f, b1, hbf);
    k_gemm2<<<dim3((NT2 + 31) / 32, M_ROWS / 64), 256, 0, stream>>>(hbf, W2f, b2, out);
}
